// RNN_70841190580309
// MI455X (gfx1250) — compile-verified, matrix operand order checked by reading, not run on hardware
//
#include <hip/hip_runtime.h>
#include <hip/hip_bf16.h>

// ---------------------------------------------------------------------------
// Seq2seq GRU (B=64,T=512,H=512) on gfx1250:
//  - bf16 WMMA (v_wmma_f32_16x16x32_bf16) for all GEMMs, fp32 accumulate
//  - persistent single-workgroup scan kernels; fp32 hidden state in REGISTERS
//    (each wave persistently owns a 16-wide hidden tile), bf16 shadow in LDS
//  - TDM tensor_load_to_lds + s_wait_tensorcnt for per-step activation tiles
//  - wave32: 32 waves/block
// ---------------------------------------------------------------------------

typedef __bf16 bf16t;
typedef __attribute__((ext_vector_type(16))) __bf16 v16bf;
typedef __attribute__((ext_vector_type(8)))  float  v8f;
typedef __attribute__((ext_vector_type(4)))  unsigned int u32x4;
typedef __attribute__((ext_vector_type(8)))  unsigned int u32x8;

#define BATCH 64
#define TSEQ  512
#define HIDD  512
#define GATES 1536
#define OUTD  256

#define WMMA_BF16(A, B, C) \
  __builtin_amdgcn_wmma_f32_16x16x32_bf16(false, (A), false, (B), (short)0, (C), false, false)

// ---- WMMA fragment loaders -------------------------------------------------
// 16-bit A 16x32 layout (ISA 7.12.2): lanes 0-15 rows M=0..15 hold K = k0+0..7
// (VGPR0-3) and k0+16..23 (VGPR4-7); lanes 16-31 hold K = k0+8..15 / k0+24..31.
static __device__ inline v16bf loadA_bf16(const bf16t* base, int ld, int m0, int k0, int lane) {
  int row = m0 + (lane & 15);
  int off = (lane < 16) ? 0 : 8;
  const bf16t* p = base + (size_t)row * ld + k0 + off;
  v16bf a;
#pragma unroll
  for (int i = 0; i < 8; ++i) { a[i] = p[i]; a[8 + i] = p[16 + i]; }
  return a;
}

static __device__ inline v16bf loadA_f32(const float* base, int ld, int m0, int k0, int lane) {
  int row = m0 + (lane & 15);
  int off = (lane < 16) ? 0 : 8;
  const float* p = base + (size_t)row * ld + k0 + off;
  v16bf a;
#pragma unroll
  for (int i = 0; i < 8; ++i) { a[i] = (bf16t)p[i]; a[8 + i] = (bf16t)p[16 + i]; }
  return a;
}

// B matrix is W^T (KxN) with W row-major (N,K): lane owns column n = n0+lane%16,
// i.e. row n of W, reading contiguous K runs -> coalesced b128 loads.
static __device__ inline v16bf loadB_w(const bf16t* W, int ldk, int n0, int k0, int lane) {
  int n = n0 + (lane & 15);
  int off = (lane < 16) ? 0 : 8;
  const bf16t* p = W + (size_t)n * ldk + k0 + off;
  v16bf b;
#pragma unroll
  for (int i = 0; i < 8; ++i) { b[i] = p[i]; b[8 + i] = p[16 + i]; }
  return b;
}

static __device__ inline float sigmoidf_(float x) { return 1.0f / (1.0f + __expf(-x)); }

// ---- TDM: DMA one 2D bf16 tile (rows x cols, row stride) into LDS ----------
// D# group0/group1 per ISA 8.3/8.4; 2D tile -> 2-SGPR-group form of
// tensor_load_to_lds. Issued by a single wave; EXEC is ignored by TDM.
static __device__ inline void tdm_load_tile_bf16(const bf16t* gsrc, unsigned lds_off,
                                                 int cols, int rows, int row_stride_elems) {
  unsigned long long ga = (unsigned long long)(const void*)gsrc;
  u32x4 g0;
  g0[0] = 1u;                                    // count=1, user descriptor
  g0[1] = lds_off;                               // lds_addr (bytes)
  g0[2] = (unsigned)ga;                          // global_addr[31:0]
  g0[3] = (unsigned)(ga >> 32) | (2u << 30);     // global_addr[56:32] | type=2
  unsigned td0 = (unsigned)cols;                 // tensor_dim0 (elems)
  unsigned td1 = (unsigned)(BATCH * TSEQ);       // tensor_dim1 (lines, generous)
  unsigned st0 = (unsigned)row_stride_elems;     // tensor_dim0_stride (elems)
  u32x8 g1;
  g1[0] = 1u << 16;                              // workgroup_mask=0, data_size=1 (2B)
  g1[1] = (td0 & 0xffffu) << 16;                 // [47:32]=bar_addr 0, [63:48]=td0.lo
  g1[2] = (td0 >> 16) | ((td1 & 0xffffu) << 16); // td0.hi | td1.lo
  g1[3] = (td1 >> 16) | (((unsigned)cols & 0xffffu) << 16);  // td1.hi | tile_dim0
  g1[4] = (unsigned)rows;                        // tile_dim1 | tile_dim2=0
  g1[5] = st0;                                   // tensor_dim0_stride[31:0]
  g1[6] = 0u;                                    // stride[47:32] | dim1_stride.lo
  g1[7] = 0u;
  asm volatile("tensor_load_to_lds %0, %1" :: "s"(g0), "s"(g1) : "memory");
}

// ---- fp32 -> bf16 conversion ----------------------------------------------
__global__ void cvt_f32_bf16(const float* __restrict__ src, bf16t* __restrict__ dst, long n) {
  long i = (long)blockIdx.x * blockDim.x + threadIdx.x;
  long stride = (long)gridDim.x * blockDim.x;
  for (; i < n; i += stride) dst[i] = (bf16t)src[i];
}

// ---- teacher-forcing mask: threefry2x32 (JAX default PRNG, key(1234)) ------
static __device__ inline unsigned rotl32(unsigned x, int r) { return (x << r) | (x >> (32 - r)); }

__global__ void tf_mask_kernel(const int* epoch, const int* total_epochs,
                               int* __restrict__ mask, float* __restrict__ ptf_out) {
  int i = threadIdx.x;  // 256 threads; thread i produces bits[i] and bits[i+256]
  float e  = (float)epoch[0];
  float te = (float)total_epochs[0];
  float p  = te / (te + __expf(e / te));
  if (i == 0) *ptf_out = p;

  unsigned ks0 = 0u, ks1 = 1234u;                 // key(1234) = [0, 1234]
  unsigned ks2 = ks0 ^ ks1 ^ 0x1BD11BDAu;
  unsigned x0 = (unsigned)i + ks0;                // counts[:256]
  unsigned x1 = (unsigned)(i + 256) + ks1;        // counts[256:]
  const int rotA[4] = {13, 15, 26, 6};
  const int rotB[4] = {17, 29, 16, 24};
#pragma unroll
  for (int g = 0; g < 5; ++g) {
    const int* rot = (g & 1) ? rotB : rotA;
#pragma unroll
    for (int r = 0; r < 4; ++r) { x0 += x1; x1 = rotl32(x1, rot[r]); x1 ^= x0; }
    switch (g) {
      case 0: x0 += ks1; x1 += ks2 + 1u; break;
      case 1: x0 += ks2; x1 += ks0 + 2u; break;
      case 2: x0 += ks0; x1 += ks1 + 3u; break;
      case 3: x0 += ks1; x1 += ks2 + 4u; break;
      case 4: x0 += ks2; x1 += ks0 + 5u; break;
    }
  }
  float u0 = __uint_as_float((x0 >> 9) | 0x3f800000u) - 1.0f;
  float u1 = __uint_as_float((x1 >> 9) | 0x3f800000u) - 1.0f;
  mask[i]       = (u0 < p) ? 1 : 0;
  mask[i + 256] = (u1 < p) ? 1 : 0;
}

// ---- persistent GRU scan (one block per direction) -------------------------
struct GruDir {
  const float* x_f32;   // fp32 input (layer 0) or null
  const bf16t* x_bf;    // bf16 input (layer 1) or null
  const bf16t* Wih;     // (1536, Din) bf16
  const bf16t* Whh;     // (1536, 512) bf16
  const float* bih;     // (1536)
  const float* bhh;     // (1536)
  bf16t*       out;     // (B, T, out_stride) bf16 at column out_col, or null
  int          out_col;
  float*       final_h; // (B, 512) fp32, or null
  int          reverse;
};

__global__ __launch_bounds__(1024) void gru_scan_kernel(GruDir d0, GruDir d1,
                                                        int Din, int out_stride) {
  GruDir d = (blockIdx.x == 0) ? d0 : d1;
  extern __shared__ char smem[];
  bf16t* hbf = (bf16t*)smem;                                   // 64x512 bf16 (64 KB)
  bf16t* xs  = (bf16t*)(smem + BATCH * HIDD * sizeof(bf16t));  // 64xDin bf16

  const int tid  = threadIdx.x;
  const int lane = tid & 31;
  const int wave = tid >> 5;

  for (int i = tid; i < BATCH * HIDD; i += blockDim.x) hbf[i] = (bf16t)0.0f;
  __syncthreads();

  // wave owns hidden tile j = wave (cols wave*16 .. wave*16+15); fp32 hidden
  // state for those elements lives in registers across the whole scan.
  const int jr = wave * 16;
  const int jz = HIDD + wave * 16;
  const int jn = 2 * HIDD + wave * 16;
  const int col = wave * 16 + (lane & 15);
  const float bias_r  = d.bih[col] + d.bhh[col];
  const float bias_z  = d.bih[HIDD + col] + d.bhh[HIDD + col];
  const float bias_xn = d.bih[2 * HIDD + col];
  const float bias_hn = d.bhh[2 * HIDD + col];

  v8f hprev[4];
#pragma unroll
  for (int m = 0; m < 4; ++m)
#pragma unroll
    for (int v = 0; v < 8; ++v) hprev[m][v] = 0.0f;

  const unsigned xs_lds_off = (unsigned)((const char*)xs - (const char*)smem);

  for (int t = 0; t < TSEQ; ++t) {
    const int tt = d.reverse ? (TSEQ - 1 - t) : t;

    // stage x[:, tt, :] into LDS as bf16
    if (d.x_bf) {
      // Tensor Data Mover: one wave DMAs the 64 x Din tile (row stride T*Din)
      if (wave == 0) {
        tdm_load_tile_bf16(d.x_bf + (size_t)tt * Din, xs_lds_off,
                           Din, BATCH, TSEQ * Din);
        __builtin_amdgcn_s_wait_tensorcnt(0);
      }
    } else {
      const int ttn = d.reverse ? (tt > 0 ? tt - 1 : 0) : (tt < TSEQ - 1 ? tt + 1 : tt);
      for (int i = tid * 4; i < BATCH * Din; i += blockDim.x * 4) {
        int b = i / Din, k = i - b * Din;
        float4 v = *(const float4*)(d.x_f32 + ((size_t)b * TSEQ + tt) * Din + k);
        xs[i] = (bf16t)v.x; xs[i + 1] = (bf16t)v.y;
        xs[i + 2] = (bf16t)v.z; xs[i + 3] = (bf16t)v.w;
        __builtin_prefetch(d.x_f32 + ((size_t)b * TSEQ + ttn) * Din + k, 0, 0);
      }
    }
    __syncthreads();

#pragma unroll
    for (int m = 0; m < 4; ++m) {
      v8f ar = {}, az = {}, axn = {}, ahn = {};
      // x-part: gx_r, gx_z, gx_n   (r/z share accumulators with h-part)
      for (int k0 = 0; k0 < Din; k0 += 32) {
        v16bf A = loadA_bf16(xs, Din, m * 16, k0, lane);
        ar  = WMMA_BF16(A, loadB_w(d.Wih, Din, jr, k0, lane), ar);
        az  = WMMA_BF16(A, loadB_w(d.Wih, Din, jz, k0, lane), az);
        axn = WMMA_BF16(A, loadB_w(d.Wih, Din, jn, k0, lane), axn);
      }
      // h-part from bf16 shadow: gh_r, gh_z merged; gh_n separate (r * hn)
      for (int k0 = 0; k0 < HIDD; k0 += 32) {
        v16bf A = loadA_bf16(hbf, HIDD, m * 16, k0, lane);
        ar  = WMMA_BF16(A, loadB_w(d.Whh, HIDD, jr, k0, lane), ar);
        az  = WMMA_BF16(A, loadB_w(d.Whh, HIDD, jz, k0, lane), az);
        ahn = WMMA_BF16(A, loadB_w(d.Whh, HIDD, jn, k0, lane), ahn);
      }
#pragma unroll
      for (int v = 0; v < 8; ++v) {
        float r = sigmoidf_(ar[v] + bias_r);
        float z = sigmoidf_(az[v] + bias_z);
        float n = tanhf(axn[v] + bias_xn + r * (ahn[v] + bias_hn));
        hprev[m][v] = (1.0f - z) * n + z * hprev[m][v];
      }
    }
    __syncthreads();  // all GEMM reads of hbf (and xs) complete
#pragma unroll
    for (int m = 0; m < 4; ++m)
#pragma unroll
      for (int v = 0; v < 8; ++v) {
        int row = m * 16 + v + ((lane < 16) ? 0 : 8);
        bf16t hv = (bf16t)hprev[m][v];
        hbf[row * HIDD + col] = hv;
        if (d.out)
          d.out[((size_t)row * TSEQ + tt) * out_stride + d.out_col + col] = hv;
      }
    __syncthreads();
  }

  if (d.final_h) {
#pragma unroll
    for (int m = 0; m < 4; ++m)
#pragma unroll
      for (int v = 0; v < 8; ++v) {
        int row = m * 16 + v + ((lane < 16) ? 0 : 8);
        d.final_h[row * HIDD + col] = hprev[m][v];
      }
  }
}

// ---- decoder scan: GRU cell + LayerNorm + FC + scheduled sampling ----------
__global__ __launch_bounds__(1024) void decoder_kernel(
    const float* __restrict__ hf, const float* __restrict__ hb,   // (64,512) each
    const bf16t* __restrict__ dW, const bf16t* __restrict__ dU,   // (1536,1280),(1536,512)
    const float* __restrict__ dbi, const float* __restrict__ dbh,
    const float* __restrict__ ln_g, const float* __restrict__ ln_b,
    const bf16t* __restrict__ fcW, const float* __restrict__ fcb, // (256,512)
    const float* __restrict__ gt,                                 // (64,512,256)
    const int* __restrict__ mask,
    float* __restrict__ ctx_gates,                                // (64,1536) ws
    float* __restrict__ poses)                                    // (64,512,256)
{
  extern __shared__ char smem[];
  float* h    = (float*)smem;                                         // 128 KB (for LN)
  bf16t* hbf  = (bf16t*)(smem + BATCH * HIDD * 4);                    //  64 KB (GEMM A)
  bf16t* prev = (bf16t*)(smem + BATCH * HIDD * 4 + BATCH * HIDD * 2); //  32 KB
  bf16t* nrm  = (bf16t*)(smem + BATCH * HIDD * 4 + BATCH * HIDD * 2
                              + BATCH * OUTD * 2);                    //  64 KB

  const int tid  = threadIdx.x;
  const int lane = tid & 31;
  const int wave = tid >> 5;
  const int LDW  = OUTD + 2 * HIDD;  // 1280, dec_Wih leading K

  for (int i = tid; i < BATCH * HIDD; i += blockDim.x) { h[i] = 0.0f; hbf[i] = (bf16t)0.0f; }
  for (int i = tid; i < BATCH * OUTD; i += blockDim.x) {
    int b = i >> 8, c = i & 255;
    prev[i] = (bf16t)gt[((size_t)b * TSEQ + 0) * OUTD + c];  // initial prev = gt[:,0,:]
  }

  // phase 0 (once): ctx_gates = dbi + h_ctx @ dW[:,256:]^T   (step-invariant)
  for (int c = wave; c < 4 * 96; c += 32) {
    int m = c / 96, n = c % 96;
    v8f acc = {};
    for (int k0 = 0; k0 < 2 * HIDD; k0 += 32) {
      const float* src = (k0 < HIDD) ? hf : hb;
      int kk = (k0 < HIDD) ? k0 : (k0 - HIDD);
      v16bf A = loadA_f32(src, HIDD, m * 16, kk, lane);
      acc = WMMA_BF16(A, loadB_w(dW, LDW, n * 16, OUTD + k0, lane), acc);
    }
    int gcol = n * 16 + (lane & 15);
    float bi = dbi[gcol];
#pragma unroll
    for (int v = 0; v < 8; ++v) {
      int row = m * 16 + v + ((lane < 16) ? 0 : 8);
      ctx_gates[(size_t)row * GATES + gcol] = acc[v] + bi;
    }
  }
  __threadfence_block();
  __syncthreads();

  const int jr = wave * 16;
  const int jz = HIDD + wave * 16;
  const int jn = 2 * HIDD + wave * 16;
  const int col = wave * 16 + (lane & 15);
  const float bhr = dbh[col], bhz = dbh[HIDD + col], bhn = dbh[2 * HIDD + col];

  v8f hprev[4];
#pragma unroll
  for (int m = 0; m < 4; ++m)
#pragma unroll
    for (int v = 0; v < 8; ++v) hprev[m][v] = 0.0f;

  for (int t = 0; t < TSEQ; ++t) {
#pragma unroll
    for (int m = 0; m < 4; ++m) {
      v8f ar, az, axn, ahn = {};
#pragma unroll
      for (int v = 0; v < 8; ++v) {               // seed with precomputed ctx gates
        int row = m * 16 + v + ((lane < 16) ? 0 : 8);
        ar[v]  = ctx_gates[(size_t)row * GATES + col];
        az[v]  = ctx_gates[(size_t)row * GATES + HIDD + col];
        axn[v] = ctx_gates[(size_t)row * GATES + 2 * HIDD + col];
      }
      for (int k0 = 0; k0 < OUTD; k0 += 32) {     // prev @ dW[:, :256]^T
        v16bf A = loadA_bf16(prev, OUTD, m * 16, k0, lane);
        ar  = WMMA_BF16(A, loadB_w(dW, LDW, jr, k0, lane), ar);
        az  = WMMA_BF16(A, loadB_w(dW, LDW, jz, k0, lane), az);
        axn = WMMA_BF16(A, loadB_w(dW, LDW, jn, k0, lane), axn);
      }
      for (int k0 = 0; k0 < HIDD; k0 += 32) {     // h @ dU^T (bf16 shadow)
        v16bf A = loadA_bf16(hbf, HIDD, m * 16, k0, lane);
        ar  = WMMA_BF16(A, loadB_w(dU, HIDD, jr, k0, lane), ar);
        az  = WMMA_BF16(A, loadB_w(dU, HIDD, jz, k0, lane), az);
        ahn = WMMA_BF16(A, loadB_w(dU, HIDD, jn, k0, lane), ahn);
      }
#pragma unroll
      for (int v = 0; v < 8; ++v) {
        float r = sigmoidf_(ar[v] + bhr);
        float z = sigmoidf_(az[v] + bhz);
        float n = tanhf(axn[v] + r * (ahn[v] + bhn));
        hprev[m][v] = (1.0f - z) * n + z * hprev[m][v];
      }
    }
    __syncthreads();
#pragma unroll
    for (int m = 0; m < 4; ++m)
#pragma unroll
      for (int v = 0; v < 8; ++v) {
        int row = m * 16 + v + ((lane < 16) ? 0 : 8);
        float hv = hprev[m][v];
        h[row * HIDD + col] = hv;
        hbf[row * HIDD + col] = (bf16t)hv;
      }
    __syncthreads();

    // LayerNorm: wave handles rows {wave, wave+32}, wave32 shuffle reduction
#pragma unroll
    for (int rr = 0; rr < 2; ++rr) {
      int row = wave + rr * 32;
      float s = 0.0f, s2 = 0.0f;
#pragma unroll
      for (int i2 = 0; i2 < 16; ++i2) {
        float v = h[row * HIDD + lane + i2 * 32];
        s += v; s2 += v * v;
      }
#pragma unroll
      for (int o = 16; o > 0; o >>= 1) {
        s  += __shfl_xor(s, o, 32);
        s2 += __shfl_xor(s2, o, 32);
      }
      float mu   = s * (1.0f / HIDD);
      float var  = s2 * (1.0f / HIDD) - mu * mu;
      float rstd = rsqrtf(var + 1e-5f);
#pragma unroll
      for (int i2 = 0; i2 < 16; ++i2) {
        int c2 = lane + i2 * 32;
        float v = h[row * HIDD + c2];
        nrm[row * HIDD + c2] = (bf16t)((v - mu) * rstd * ln_g[c2] + ln_b[c2]);
      }
    }
    __syncthreads();

    // FC: pose = nrm @ fcW^T + fcb; scheduled sampling for next prev
    int tf = mask[t];
#pragma unroll
    for (int cc = 0; cc < 2; ++cc) {
      int c = wave * 2 + cc;
      int m = c >> 4, n = c & 15;
      v8f acc = {};
      for (int k0 = 0; k0 < HIDD; k0 += 32) {
        v16bf A = loadA_bf16(nrm, HIDD, m * 16, k0, lane);
        acc = WMMA_BF16(A, loadB_w(fcW, HIDD, n * 16, k0, lane), acc);
      }
      int pc = n * 16 + (lane & 15);
      float bb = fcb[pc];
#pragma unroll
      for (int v = 0; v < 8; ++v) {
        int row = m * 16 + v + ((lane < 16) ? 0 : 8);
        float pose = acc[v] + bb;
        poses[((size_t)row * TSEQ + t) * OUTD + pc] = pose;
        float nxt = tf ? gt[((size_t)row * TSEQ + t) * OUTD + pc] : pose;
        prev[row * OUTD + pc] = (bf16t)nxt;
      }
    }
    __syncthreads();
  }
}

// ---------------------------------------------------------------------------
extern "C" void kernel_launch(void* const* d_in, const int* in_sizes, int n_in,
                              void* d_out, int out_size, void* d_ws, size_t ws_size,
                              hipStream_t stream) {
  (void)in_sizes; (void)n_in; (void)out_size; (void)ws_size;
  const float* gloss = (const float*)d_in[0];
  const float* gt    = (const float*)d_in[1];
  const int*   epoch = (const int*)d_in[2];
  const int*   tot   = (const int*)d_in[3];

  // workspace carve-out (256B aligned)
  size_t off = 0;
  auto take = [&](size_t bytes) -> void* {
    void* p = (char*)d_ws + off;
    off += (bytes + 255) & ~(size_t)255;
    return p;
  };
  bf16t* wih0f = (bf16t*)take((size_t)GATES * 512 * 2);
  bf16t* whh0f = (bf16t*)take((size_t)GATES * 512 * 2);
  bf16t* wih0b = (bf16t*)take((size_t)GATES * 512 * 2);
  bf16t* whh0b = (bf16t*)take((size_t)GATES * 512 * 2);
  bf16t* wih1f = (bf16t*)take((size_t)GATES * 1024 * 2);
  bf16t* whh1f = (bf16t*)take((size_t)GATES * 512 * 2);
  bf16t* wih1b = (bf16t*)take((size_t)GATES * 1024 * 2);
  bf16t* whh1b = (bf16t*)take((size_t)GATES * 512 * 2);
  bf16t* dWbf  = (bf16t*)take((size_t)GATES * 1280 * 2);
  bf16t* dUbf  = (bf16t*)take((size_t)GATES * 512 * 2);
  bf16t* fcWbf = (bf16t*)take((size_t)OUTD * 512 * 2);
  bf16t* x1    = (bf16t*)take((size_t)BATCH * TSEQ * 1024 * 2);  // concat(of,ob) bf16
  float* hfbuf = (float*)take((size_t)BATCH * HIDD * 4);
  float* hbbuf = (float*)take((size_t)BATCH * HIDD * 4);
  float* ctx   = (float*)take((size_t)BATCH * GATES * 4);
  int*   mask  = (int*)take((size_t)TSEQ * 4);

  float* poses = (float*)d_out;
  float* ptf   = poses + (size_t)BATCH * TSEQ * OUTD;

  // 1) teacher-forcing mask + p_tf (threefry2x32)
  tf_mask_kernel<<<1, 256, 0, stream>>>(epoch, tot, mask, ptf);

  // 2) weight conversions fp32 -> bf16
  auto cvt = [&](const void* s, bf16t* dst, long n) {
    int blocks = (int)((n + 256 * 8 - 1) / (256 * 8));
    if (blocks > 1024) blocks = 1024;
    cvt_f32_bf16<<<blocks, 256, 0, stream>>>((const float*)s, dst, n);
  };
  cvt(d_in[4],  wih0f, (long)GATES * 512);
  cvt(d_in[5],  whh0f, (long)GATES * 512);
  cvt(d_in[8],  wih0b, (long)GATES * 512);
  cvt(d_in[9],  whh0b, (long)GATES * 512);
  cvt(d_in[12], wih1f, (long)GATES * 1024);
  cvt(d_in[13], whh1f, (long)GATES * 512);
  cvt(d_in[16], wih1b, (long)GATES * 1024);
  cvt(d_in[17], whh1b, (long)GATES * 512);
  cvt(d_in[20], dWbf,  (long)GATES * 1280);
  cvt(d_in[21], dUbf,  (long)GATES * 512);
  cvt(d_in[26], fcWbf, (long)OUTD * 512);

  // allow >64KB dynamic LDS (up to 320KB per workgroup on CDNA5)
  size_t smem0 = (size_t)BATCH * HIDD * 2 + (size_t)BATCH * 512 * 2;   // 128 KB
  size_t smem1 = (size_t)BATCH * HIDD * 2 + (size_t)BATCH * 1024 * 2;  // 192 KB
  size_t smem2 = (size_t)BATCH * HIDD * 4 + (size_t)BATCH * HIDD * 2
               + (size_t)BATCH * OUTD * 2 + (size_t)BATCH * HIDD * 2;  // 288 KB
  hipFuncSetAttribute(reinterpret_cast<const void*>(&gru_scan_kernel),
                      hipFuncAttributeMaxDynamicSharedMemorySize, (int)smem1);
  hipFuncSetAttribute(reinterpret_cast<const void*>(&decoder_kernel),
                      hipFuncAttributeMaxDynamicSharedMemorySize, (int)smem2);

  // 3) encoder layer 0 (bidirectional, fp32 input) -> x1 bf16 (B,T,1024)
  GruDir f0 = {gloss, nullptr, wih0f, whh0f, (const float*)d_in[6], (const float*)d_in[7],
               x1, 0, nullptr, 0};
  GruDir b0 = {gloss, nullptr, wih0b, whh0b, (const float*)d_in[10], (const float*)d_in[11],
               x1, 512, nullptr, 1};
  gru_scan_kernel<<<2, 1024, smem0, stream>>>(f0, b0, 512, 1024);

  // 4) encoder layer 1 (bidirectional, bf16 input via TDM, only final hiddens)
  GruDir f1 = {nullptr, x1, wih1f, whh1f, (const float*)d_in[14], (const float*)d_in[15],
               nullptr, 0, hfbuf, 0};
  GruDir b1 = {nullptr, x1, wih1b, whh1b, (const float*)d_in[18], (const float*)d_in[19],
               nullptr, 0, hbbuf, 1};
  gru_scan_kernel<<<2, 1024, smem1, stream>>>(f1, b1, 1024, 0);

  // 5) decoder scan (GRU + LayerNorm + FC + scheduled sampling)
  decoder_kernel<<<1, 1024, smem2, stream>>>(
      hfbuf, hbbuf, dWbf, dUbf,
      (const float*)d_in[22], (const float*)d_in[23],
      (const float*)d_in[24], (const float*)d_in[25],
      fcWbf, (const float*)d_in[27], gt, mask, ctx, poses);
}
